// SparseLinearSuper_2645699855062
// MI455X (gfx1250) — compile-verified, matrix operand order checked by reading, not run on hardware
//
#include <hip/hip_runtime.h>

// ---------------------------------------------------------------------------
// SparseLinearSuper (2:4 structured-sparse linear) for MI455X / gfx1250.
//   out[M,O] = x[M,K] * prune24(W)[O,K]^T + bias    (M=65536, O=K=768)
//
// Primary path: V_SWMMAC_F32_16X16X64_BF16 (2:4 sparse A operand = weights,
// 2x dense matrix rate, half the weight traffic).  Orientation:
//   out^T = W_sparse * x^T
//   A = compressed W (16x32 bf16 stored -> 16x64 expanded) + 2-bit indices
//   B = x tile (64x16 bf16, dense, per-lane contiguous K runs)
//   D = 16(o) x 16(m) f32 -> per-lane outputs contiguous in o (float4 stores)
// W/idx tiles staged with GLOBAL_LOAD_ASYNC_TO_LDS (ASYNCcnt), x staged via
// regs with branchless f32->bf16 convert, +global_prefetch on the x stream.
// Dense v_wmma_f32_16x16x32_bf16 kernel kept below as fallback.
// ---------------------------------------------------------------------------

typedef __attribute__((ext_vector_type(32))) __bf16 v32bf;
typedef __attribute__((ext_vector_type(16))) __bf16 v16bf;
typedef __attribute__((ext_vector_type(8)))  __bf16 v8bf;
typedef __attribute__((ext_vector_type(8)))  float  v8f;

#define THREADS 256     // 8 wave32s

// ---- sparse path tiling ----
#define BO  256         // output-channel block (A rows)
#define BMX 128         // x-row block (B cols)
#define WPITCH 40       // LDS pitch for compressed W rows (32 data + 8 pad bf16)
#define XPITCH 72       // LDS pitch for x rows (64 data + 8 pad bf16)
#define WS_HALF (BO * WPITCH)        // ushorts per buffer
#define XS_HALF (BMX * XPITCH)       // ushorts per buffer
#define IS_HALF (BO * 2)             // uints per buffer
// byte offsets inside the (otherwise empty) dynamic-LDS block
#define WS_OFF 0
#define XS_OFF (2 * WS_HALF * 2)
#define IS_OFF (XS_OFF + 2 * XS_HALF * 2)
#define SMEM_SPARSE (IS_OFF + 2 * IS_HALF * 4)

// ---- dense path tiling (fallback kernel) ----
#define BM 128
#define BN 256
#define BK 64
#define LDP 72

// Branchless f32 -> bf16 (round-to-nearest-even, NaN quieted via v_cndmask,
// no EXEC-divergent control flow in the hot loop).
__device__ __forceinline__ unsigned int f32_to_bf16_bits(float f) {
  unsigned int u = __float_as_uint(f);
  unsigned int rne  = (u + 0x7fffu + ((u >> 16) & 1u)) >> 16;
  unsigned int nanv = (u >> 16) | 0x0040u;
  bool isnan = (u & 0x7fffffffu) > 0x7f800000u;
  return (isnan ? nanv : rne) & 0xffffu;
}

// Async global -> LDS copies (CDNA5, tracked by ASYNCcnt).  lds_off is the
// wave-relative LDS byte offset (dynamic LDS starts at 0 in these kernels).
__device__ __forceinline__ void async_load_b128(unsigned lds_off,
                                                const void* gaddr) {
  asm volatile("global_load_async_to_lds_b128 %0, %1, off"
               :: "v"(lds_off), "v"((unsigned long long)gaddr)
               : "memory");
}
__device__ __forceinline__ void async_load_b32(unsigned lds_off,
                                               const void* gaddr) {
  asm volatile("global_load_async_to_lds_b32 %0, %1, off"
               :: "v"(lds_off), "v"((unsigned long long)gaddr)
               : "memory");
}
__device__ __forceinline__ void wait_async0() {
  asm volatile("s_wait_asynccnt 0" ::: "memory");
}

// ---------------------------------------------------------------------------
// Prep: per (o, 64-K block) produce 32 compressed bf16 + two 32-bit index
// words in the ISA 7.12.4 16-bit-sparse layout (idx0 < idx1, stable-argsort
// tie semantics identical to the reference).
// ---------------------------------------------------------------------------
__global__ void prep_weight_sparse24(const float* __restrict__ w,
                                     unsigned short* __restrict__ wc,  // [O][K/2]
                                     unsigned int* __restrict__ widx,  // [O][K/64][2]
                                     int O, int K) {
  int nkb = K >> 6;
  int t = blockIdx.x * blockDim.x + threadIdx.x;
  if (t >= O * nkb) return;
  int o = t / nkb, kb = t % nkb;
  const float* src = w + (size_t)o * K + kb * 64;

  unsigned int cv[16];        // 32 bf16 packed as 16 uints
  unsigned int idxw[2] = {0u, 0u};
  #pragma unroll
  for (int g = 0; g < 16; ++g) {
    float4 v4 = *(const float4*)(src + g * 4);
    float v[4] = {v4.x, v4.y, v4.z, v4.w};
    float a[4] = {fabsf(v4.x), fabsf(v4.y), fabsf(v4.z), fabsf(v4.w)};
    int i0 = 0;                                    // first (stable) min
    #pragma unroll
    for (int q = 1; q < 4; ++q) if (a[q] < a[i0]) i0 = q;
    int i1 = -1;                                   // first min of the rest
    #pragma unroll
    for (int q = 0; q < 4; ++q) {
      if (q == i0) continue;
      if (i1 < 0 || a[q] < a[i1]) i1 = q;
    }
    int j0 = -1, j1 = -1;                          // kept pair, ascending
    #pragma unroll
    for (int q = 0; q < 4; ++q) {
      if (q == i0 || q == i1) continue;
      if (j0 < 0) j0 = q; else j1 = q;
    }
    cv[g] = f32_to_bf16_bits(v[j0]) | (f32_to_bf16_bits(v[j1]) << 16);
    int h = g >> 3, f = g & 7;
    idxw[h] |= ((unsigned)j0 << (4 * f)) | ((unsigned)j1 << (4 * f + 2));
  }
  uint4* dst = (uint4*)(wc + (size_t)o * (K >> 1) + kb * 32);
  dst[0] = make_uint4(cv[0], cv[1], cv[2], cv[3]);
  dst[1] = make_uint4(cv[4], cv[5], cv[6], cv[7]);
  dst[2] = make_uint4(cv[8], cv[9], cv[10], cv[11]);
  dst[3] = make_uint4(cv[12], cv[13], cv[14], cv[15]);
  widx[(size_t)t * 2 + 0] = idxw[0];
  widx[(size_t)t * 2 + 1] = idxw[1];
}

// ---------------------------------------------------------------------------
// Sparse GEMM: out^T = W_sparse * x^T via v_swmmac_f32_16x16x64_bf16.
// ---------------------------------------------------------------------------
__global__ __launch_bounds__(THREADS) void sparse_linear_swmmac(
    const float* __restrict__ x, const unsigned short* __restrict__ wc,
    const unsigned int* __restrict__ widx, const float* __restrict__ bias,
    float* __restrict__ out, int M, int O, int K, int nkb) {
  extern __shared__ unsigned char smem_raw[];
  unsigned short* Ws = (unsigned short*)(smem_raw + WS_OFF); // [2][BO][WPITCH]
  unsigned short* Xs = (unsigned short*)(smem_raw + XS_OFF); // [2][BMX][XPITCH]
  unsigned int*   Is = (unsigned int*)(smem_raw + IS_OFF);   // [2][BO][2]

  const int tid  = threadIdx.x;
  const int lane = tid & 31;                 // wave32
  const int wave = tid >> 5;                 // 0..7
  const int wo   = wave & 3;                 // 0..3 : 64-wide o band
  const int wmx  = wave >> 2;                // 0..1 : 64-wide m band
  const int mrow = lane & 15;
  const int half = lane >> 4;                // 0/1
  const int hsel = half * 8;

  const int blockO = blockIdx.x * BO;
  const int blockM = blockIdx.y * BMX;

  const v8f zero8 = {0.f, 0.f, 0.f, 0.f, 0.f, 0.f, 0.f, 0.f};
  v8f acc[4][4];
  #pragma unroll
  for (int i = 0; i < 4; ++i)
    #pragma unroll
    for (int j = 0; j < 4; ++j) acc[i][j] = zero8;

  float4 xreg[8];   // x tile (f32, converted at LDS store)

// Issue async W + index copies straight into LDS buffer BUF (ASYNCcnt).
#define S_ASYNC_W(KB, BUF)                                                     \
  {                                                                            \
    _Pragma("unroll")                                                          \
    for (int i = 0; i < 4; ++i) {         /* 1024 x 16B : BO x 4 */            \
      int slot = tid + i * THREADS;                                            \
      int o = slot >> 2, c16 = slot & 3;                                       \
      int go = blockO + o;                                                     \
      if (go < O)                                                              \
        async_load_b128(                                                       \
            WS_OFF + ((BUF) * WS_HALF + o * WPITCH + c16 * 8) * 2u,            \
            wc + (size_t)go * (K >> 1) + (KB) * 32 + c16 * 8);                 \
    }                                                                          \
    _Pragma("unroll")                                                          \
    for (int i = 0; i < 2; ++i) {         /* 512 x b32 : BO x 2 */             \
      int slot = tid + i * THREADS;                                            \
      int o = slot >> 1, h = slot & 1;                                         \
      int go = blockO + o;                                                     \
      if (go < O)                                                              \
        async_load_b32(IS_OFF + ((BUF) * IS_HALF + o * 2 + h) * 4u,            \
                       widx + ((size_t)go * nkb + (KB)) * 2 + h);              \
    }                                                                          \
  }

#define S_LOAD_X(KB)                                                           \
  {                                                                            \
    _Pragma("unroll")                                                          \
    for (int i = 0; i < 8; ++i) {         /* 2048 x float4 : BMX x 16 */       \
      int slot = tid + i * THREADS;                                            \
      int m = slot >> 4, c4 = slot & 15;                                       \
      int gm = blockM + m;                                                     \
      const float* gp = x + (size_t)gm * K + (KB) * 64 + c4 * 4;               \
      xreg[i] = (gm < M) ? *(const float4*)gp                                  \
                         : make_float4(0.f, 0.f, 0.f, 0.f);                    \
      if ((KB) + 2 < nkb && gm < M)        /* HBM stream prefetch, +2 stages */\
        __builtin_prefetch((const void*)(gp + 128), 0, 0);                     \
    }                                                                          \
  }

#define S_STORE_X(BUF)                                                         \
  {                                                                            \
    unsigned short* Xd = Xs + (BUF) * XS_HALF;                                 \
    _Pragma("unroll")                                                          \
    for (int i = 0; i < 8; ++i) {                                              \
      int slot = tid + i * THREADS;                                            \
      int m = slot >> 4, c4 = slot & 15;                                       \
      unsigned int lo = f32_to_bf16_bits(xreg[i].x) |                          \
                        (f32_to_bf16_bits(xreg[i].y) << 16);                   \
      unsigned int hi = f32_to_bf16_bits(xreg[i].z) |                          \
                        (f32_to_bf16_bits(xreg[i].w) << 16);                   \
      *(uint2*)(Xd + m * XPITCH + c4 * 4) = make_uint2(lo, hi);                \
    }                                                                          \
  }

  S_ASYNC_W(0, 0)
  S_LOAD_X(0)
  S_STORE_X(0)
  wait_async0();
  __syncthreads();

  for (int s = 0; s < nkb; ++s) {
    if (s + 1 < nkb) {
      S_ASYNC_W(s + 1, (s + 1) & 1)   // async -> other buffer, overlaps compute
      S_LOAD_X(s + 1)
    }

    const unsigned short* Wcur = Ws + (s & 1) * WS_HALF;
    const unsigned short* Xcur = Xs + (s & 1) * XS_HALF;
    const unsigned int*   Icur = Is + (s & 1) * IS_HALF;

    // A fragments: compressed 16x32 bf16, standard 16-bit A layout:
    // lanes 0-15 K'={0..7,16..23}, lanes 16-31 K'={8..15,24..31}.
    v16bf afr[4];
    int   aidx[4];
    #pragma unroll
    for (int i = 0; i < 4; ++i) {
      int orow = wo * 64 + i * 16 + mrow;
      const unsigned short* p = Wcur + orow * WPITCH + hsel;
      v8bf lo = *(const v8bf*)(p);
      v8bf hi = *(const v8bf*)(p + 16);
      afr[i] = __builtin_shufflevector(lo, hi, 0, 1, 2, 3, 4, 5, 6, 7, 8, 9,
                                       10, 11, 12, 13, 14, 15);
      aidx[i] = (int)Icur[orow * 2 + half];
    }
    // B fragments: 64x16 dense, per ISA 7.12.4: V0-7 K={0..15|16..31},
    // V8-15 K={32..47|48..63} by lane half -> two 32B contiguous runs.
    v32bf bfr[4];
    #pragma unroll
    for (int j = 0; j < 4; ++j) {
      const unsigned short* p =
          Xcur + (wmx * 64 + j * 16 + mrow) * XPITCH + half * 16;
      v16bf lo = *(const v16bf*)(p);
      v16bf hi = *(const v16bf*)(p + 32);
      bfr[j] = __builtin_shufflevector(
          lo, hi, 0, 1, 2, 3, 4, 5, 6, 7, 8, 9, 10, 11, 12, 13, 14, 15, 16,
          17, 18, 19, 20, 21, 22, 23, 24, 25, 26, 27, 28, 29, 30, 31);
    }
    #pragma unroll
    for (int i = 0; i < 4; ++i)
      #pragma unroll
      for (int j = 0; j < 4; ++j)
        acc[i][j] = __builtin_amdgcn_swmmac_f32_16x16x64_bf16(
            false, afr[i], false, bfr[j], acc[i][j], aidx[i], false, false);

    if (s + 1 < nkb) S_STORE_X((s + 1) & 1)
    wait_async0();        // own async W/idx copies landed in LDS
    __syncthreads();      // ... and visible to all waves
  }

  // Epilogue: D element (o = r + 8*half, m = lane&15) -> per-lane outputs are
  // contiguous in o: bias-add + two float4 stores per 16x16 tile.
  #pragma unroll
  for (int i = 0; i < 4; ++i) {
    int ob = blockO + wo * 64 + i * 16 + hsel;
    if (ob + 7 < O) {
      float4 b0 = *(const float4*)(bias + ob);
      float4 b1 = *(const float4*)(bias + ob + 4);
      #pragma unroll
      for (int j = 0; j < 4; ++j) {
        int m = blockM + wmx * 64 + j * 16 + mrow;
        if (m < M) {
          float* dst = out + (size_t)m * O + ob;
          float4 s0 = make_float4(acc[i][j][0] + b0.x, acc[i][j][1] + b0.y,
                                  acc[i][j][2] + b0.z, acc[i][j][3] + b0.w);
          float4 s1 = make_float4(acc[i][j][4] + b1.x, acc[i][j][5] + b1.y,
                                  acc[i][j][6] + b1.z, acc[i][j][7] + b1.w);
          *(float4*)dst = s0;
          *(float4*)(dst + 4) = s1;
        }
      }
    }
  }
#undef S_ASYNC_W
#undef S_LOAD_X
#undef S_STORE_X
}

// ---------------------------------------------------------------------------
// Dense-WMMA path (kept as fallback; also keeps wmma in the binary).
// ---------------------------------------------------------------------------
__global__ void prep_weight_24(const float* __restrict__ w,
                               unsigned short* __restrict__ wb, int ngroups) {
  int g = blockIdx.x * blockDim.x + threadIdx.x;
  if (g >= ngroups) return;
  float4 v4 = ((const float4*)w)[g];
  float v[4] = {v4.x, v4.y, v4.z, v4.w};
  float a[4] = {fabsf(v4.x), fabsf(v4.y), fabsf(v4.z), fabsf(v4.w)};
  int i0 = 0;
  #pragma unroll
  for (int t = 1; t < 4; ++t) if (a[t] < a[i0]) i0 = t;
  int i1 = -1;
  #pragma unroll
  for (int t = 0; t < 4; ++t) {
    if (t == i0) continue;
    if (i1 < 0 || a[t] < a[i1]) i1 = t;
  }
  v[i0] = 0.0f; v[i1] = 0.0f;
  unsigned int lo = f32_to_bf16_bits(v[0]) | (f32_to_bf16_bits(v[1]) << 16);
  unsigned int hi = f32_to_bf16_bits(v[2]) | (f32_to_bf16_bits(v[3]) << 16);
  ((uint2*)wb)[g] = make_uint2(lo, hi);
}

__global__ __launch_bounds__(THREADS) void sparse_linear_wmma(
    const float* __restrict__ x, const unsigned short* __restrict__ wb,
    const float* __restrict__ bias, float* __restrict__ out,
    int M, int N, int K) {
  extern __shared__ unsigned short smem[];
  unsigned short* As = smem;
  unsigned short* Bs = smem + 2 * BM * LDP;

  const int tid  = threadIdx.x;
  const int lane = tid & 31;
  const int wave = tid >> 5;
  const int wm   = wave >> 2;
  const int wn   = wave & 3;
  const int mrow = lane & 15;
  const int hsel = (lane >> 4) * 8;

  const int blockM = blockIdx.y * BM;
  const int blockN = blockIdx.x * BN;

  const v8f zero8 = {0.f, 0.f, 0.f, 0.f, 0.f, 0.f, 0.f, 0.f};
  v8f acc[4][4];
  #pragma unroll
  for (int i = 0; i < 4; ++i)
    #pragma unroll
    for (int j = 0; j < 4; ++j) acc[i][j] = zero8;

  float4 areg[8];
  uint4  breg[8];

#define LOAD_GLOBAL(KT)                                                        \
  {                                                                            \
    _Pragma("unroll")                                                          \
    for (int i = 0; i < 8; ++i) {                                              \
      int slot = tid + i * THREADS;                                            \
      int row = slot >> 4, c4 = slot & 15;                                     \
      int gm = blockM + row;                                                   \
      areg[i] = (gm < M)                                                       \
                    ? *(const float4*)(x + (size_t)gm * K + (KT) + c4 * 4)     \
                    : make_float4(0.f, 0.f, 0.f, 0.f);                         \
    }                                                                          \
    _Pragma("unroll")                                                          \
    for (int i = 0; i < 8; ++i) {                                              \
      int slot = tid + i * THREADS;                                            \
      int n = slot >> 3, k8 = slot & 7;                                        \
      int gn = blockN + n;                                                     \
      breg[i] = (gn < N)                                                       \
                    ? *(const uint4*)(wb + (size_t)gn * K + (KT) + k8 * 8)     \
                    : make_uint4(0u, 0u, 0u, 0u);                              \
    }                                                                          \
  }

#define STORE_LDS(BUF)                                                         \
  {                                                                            \
    unsigned short* Ad = As + (BUF) * (BM * LDP);                              \
    unsigned short* Bd = Bs + (BUF) * (BN * LDP);                              \
    _Pragma("unroll")                                                          \
    for (int i = 0; i < 8; ++i) {                                              \
      int slot = tid + i * THREADS;                                            \
      int row = slot >> 4, c4 = slot & 15;                                     \
      unsigned int lo = f32_to_bf16_bits(areg[i].x) |                          \
                        (f32_to_bf16_bits(areg[i].y) << 16);                   \
      unsigned int hi = f32_to_bf16_bits(areg[i].z) |                          \
                        (f32_to_bf16_bits(areg[i].w) << 16);                   \
      *(uint2*)(Ad + row * LDP + c4 * 4) = make_uint2(lo, hi);                 \
    }                                                                          \
    _Pragma("unroll")                                                          \
    for (int i = 0; i < 8; ++i) {                                              \
      int slot = tid + i * THREADS;                                            \
      int n = slot >> 3, k8 = slot & 7;                                        \
      *(uint4*)(Bd + n * LDP + k8 * 8) = breg[i];                              \
    }                                                                          \
  }

  const int nstages = K / BK;

  LOAD_GLOBAL(0)
  STORE_LDS(0)
  __syncthreads();

  for (int s = 0; s < nstages; ++s) {
    if (s + 1 < nstages) LOAD_GLOBAL((s + 1) * BK)

    const unsigned short* Ac = As + (s & 1) * (BM * LDP);
    const unsigned short* Bc = Bs + (s & 1) * (BN * LDP);

    #pragma unroll
    for (int ks = 0; ks < BK; ks += 32) {
      v16bf afr[4], bfr[4];
      #pragma unroll
      for (int i = 0; i < 4; ++i) {
        const unsigned short* p =
            Ac + (wm * 64 + i * 16 + mrow) * LDP + ks + hsel;
        v8bf lo = *(const v8bf*)(p);
        v8bf hi = *(const v8bf*)(p + 16);
        afr[i] = __builtin_shufflevector(lo, hi, 0, 1, 2, 3, 4, 5, 6, 7, 8, 9,
                                         10, 11, 12, 13, 14, 15);
      }
      #pragma unroll
      for (int j = 0; j < 4; ++j) {
        const unsigned short* p =
            Bc + (wn * 64 + j * 16 + mrow) * LDP + ks + hsel;
        v8bf lo = *(const v8bf*)(p);
        v8bf hi = *(const v8bf*)(p + 16);
        bfr[j] = __builtin_shufflevector(lo, hi, 0, 1, 2, 3, 4, 5, 6, 7, 8, 9,
                                         10, 11, 12, 13, 14, 15);
      }
      #pragma unroll
      for (int i = 0; i < 4; ++i)
        #pragma unroll
        for (int j = 0; j < 4; ++j)
          acc[i][j] = __builtin_amdgcn_wmma_f32_16x16x32_bf16(
              false, afr[i], false, bfr[j], (short)0, acc[i][j], false, false);
    }

    if (s + 1 < nstages) STORE_LDS((s + 1) & 1)
    __syncthreads();
  }

  float bv[4];
  #pragma unroll
  for (int j = 0; j < 4; ++j) {
    int col = blockN + wn * 64 + j * 16 + mrow;
    bv[j] = (col < N) ? bias[col] : 0.f;
  }
  #pragma unroll
  for (int i = 0; i < 4; ++i) {
    #pragma unroll
    for (int r = 0; r < 8; ++r) {
      int row = blockM + wm * 64 + i * 16 + r + hsel;
      if (row < M) {
        #pragma unroll
        for (int j = 0; j < 4; ++j) {
          int col = blockN + wn * 64 + j * 16 + mrow;
          if (col < N) out[(size_t)row * N + col] = acc[i][j][r] + bv[j];
        }
      }
    }
  }
#undef LOAD_GLOBAL
#undef STORE_LDS
}

// ---------------------------------------------------------------------------
extern "C" void kernel_launch(void* const* d_in, const int* in_sizes, int n_in,
                              void* d_out, int out_size, void* d_ws,
                              size_t ws_size, hipStream_t stream) {
  (void)n_in; (void)out_size; (void)ws_size;
  const float* x    = (const float*)d_in[0];
  const float* w    = (const float*)d_in[1];
  const float* bias = (const float*)d_in[2];
  float* out = (float*)d_out;

  const int O = in_sizes[2];            // 768
  const int K = in_sizes[1] / O;        // 768
  const int M = in_sizes[0] / K;        // 65536
  const int nkb = K / 64;               // 12

  // workspace: compressed bf16 weights then sparse index words
  unsigned short* wc = (unsigned short*)d_ws;                 // [O][K/2]
  size_t wc_bytes = ((size_t)O * (K / 2) * 2 + 255) & ~(size_t)255;
  unsigned int* widx = (unsigned int*)((char*)d_ws + wc_bytes); // [O][nkb][2]

  int nprep = O * nkb;
  prep_weight_sparse24<<<(nprep + 255) / 256, 256, 0, stream>>>(w, wc, widx, O, K);

  dim3 grid((O + BO - 1) / BO, (M + BMX - 1) / BMX);   // (3, 512)
  size_t smem = SMEM_SPARSE;                           // ~80 KB
  hipFuncSetAttribute(reinterpret_cast<const void*>(sparse_linear_swmmac),
                      hipFuncAttributeMaxDynamicSharedMemorySize, (int)smem);
  sparse_linear_swmmac<<<grid, THREADS, smem, stream>>>(x, wc, widx, bias, out,
                                                        M, O, K, nkb);
}
